// TemporalGAT_20005957665499
// MI455X (gfx1250) — matrix operand order repaired and checked by reading, not compiled, not run on hardware
//
#include <hip/hip_runtime.h>

// Problem constants (from reference)
#define NN      20000
#define TT      8
#define FIN     64
#define HEADS   4
#define FILTERS 32
#define HID     128          // HEADS*FILTERS
#define EE      320000
#define EP      (EE + NN)    // edges + self loops = 340000
#define LEAKY   0.2f

typedef __bf16 bf16_t;
typedef bf16_t        v16bf  __attribute__((ext_vector_type(16)));
typedef bf16_t        v2bf   __attribute__((ext_vector_type(2)));
typedef unsigned short v16u16 __attribute__((ext_vector_type(16)));
typedef unsigned int   v8u   __attribute__((ext_vector_type(8)));
typedef float          v8f   __attribute__((ext_vector_type(8)));

__device__ __forceinline__ unsigned short f2bf(float f) {
    unsigned u = __float_as_uint(f);
    u += 0x7FFFu + ((u >> 16) & 1u);   // round-to-nearest-even
    return (unsigned short)(u >> 16);
}
__device__ __forceinline__ unsigned flipf(float f) {
    unsigned u = __float_as_uint(f);
    return (u & 0x80000000u) ? ~u : (u | 0x80000000u);
}
__device__ __forceinline__ float unflipf(unsigned u) {
    return __uint_as_float((u & 0x80000000u) ? (u & 0x7FFFFFFFu) : ~u);
}

// Build a 16-half bf16 A-fragment from 16 f32 values (4 float4s).
__device__ __forceinline__ v16bf make_a_frag(float4 a0, float4 a1,
                                             float4 a2, float4 a3) {
#if __has_builtin(__builtin_amdgcn_cvt_pk_bf16_f32)
    union { v16bf v; v2bf p[8]; } u;
    u.p[0] = __builtin_amdgcn_cvt_pk_bf16_f32(a0.x, a0.y);
    u.p[1] = __builtin_amdgcn_cvt_pk_bf16_f32(a0.z, a0.w);
    u.p[2] = __builtin_amdgcn_cvt_pk_bf16_f32(a1.x, a1.y);
    u.p[3] = __builtin_amdgcn_cvt_pk_bf16_f32(a1.z, a1.w);
    u.p[4] = __builtin_amdgcn_cvt_pk_bf16_f32(a2.x, a2.y);
    u.p[5] = __builtin_amdgcn_cvt_pk_bf16_f32(a2.z, a2.w);
    u.p[6] = __builtin_amdgcn_cvt_pk_bf16_f32(a3.x, a3.y);
    u.p[7] = __builtin_amdgcn_cvt_pk_bf16_f32(a3.z, a3.w);
    return u.v;
#else
    v16u16 au;
    au[0]  = f2bf(a0.x); au[1]  = f2bf(a0.y); au[2]  = f2bf(a0.z); au[3]  = f2bf(a0.w);
    au[4]  = f2bf(a1.x); au[5]  = f2bf(a1.y); au[6]  = f2bf(a1.z); au[7]  = f2bf(a1.w);
    au[8]  = f2bf(a2.x); au[9]  = f2bf(a2.y); au[10] = f2bf(a2.z); au[11] = f2bf(a2.w);
    au[12] = f2bf(a3.x); au[13] = f2bf(a3.y); au[14] = f2bf(a3.z); au[15] = f2bf(a3.w);
    return __builtin_bit_cast(v16bf, au);
#endif
}

// Load one 16x32 A fragment for tile starting at row `row`, K base `kb`.
__device__ __forceinline__ v16bf load_a_frag(const float* __restrict__ arow,
                                             int kb) {
    float4 a0 = *(const float4*)(arow + kb);
    float4 a1 = *(const float4*)(arow + kb + 4);
    float4 a2 = *(const float4*)(arow + kb + 16);
    float4 a3 = *(const float4*)(arow + kb + 20);
    return make_a_frag(a0, a1, a2, a3);
}

// ---------------------------------------------------------------------------
// Pack W[HID][Fin] (f32, row-major, h = x @ W^T) into bf16 WMMA B-fragments.
// B is K(32) x N(16) per fragment.  ISA layout (16-bit B 32x16, wave32):
//   lane L: column N = L&15 ; kbase = (L>=16 ? 8 : 0)
//   dword d holds K = kbase + 2d, 2d+1 (+8 more for d>=4)
// Fragment order: frag[(kt*8 + nt)*256 + lane*8 + d]
// ---------------------------------------------------------------------------
__global__ void pack_w(const float* __restrict__ W, int Fin,
                       unsigned* __restrict__ frag) {
    int idx = blockIdx.x * blockDim.x + threadIdx.x;
    int total = (Fin / 32) * 8 * 256;
    if (idx >= total) return;
    int d    = idx & 7;
    int lane = (idx >> 3) & 31;
    int nt   = (idx >> 8) & 7;
    int kt   = idx >> 11;
    int col  = nt * 16 + (lane & 15);                 // output feature (N)
    int kb   = kt * 32 + ((lane >> 4) << 3) + ((d >> 2) << 3); // +8 upper half-wave, +8 for d>=4
    unsigned short lo = f2bf(W[col * Fin + kb + 2 * d]);
    unsigned short hi = f2bf(W[col * Fin + kb + 2 * d + 1]);
    frag[idx] = (unsigned)lo | ((unsigned)hi << 16);
}

// ---------------------------------------------------------------------------
// h[M][HID] = A[M][Fin] @ W^T via v_wmma_f32_16x16x32_bf16.
// B fragments staged once per block in LDS; each wave owns TWO 16-row tiles
// (M=32) so every B fragment feeds two back-to-back WMMAs (halves LDS traffic
// per FLOP, hides ds latency).  8 column tiles (HID=128); K loop over Fin/32.
// A fragment (16-bit A 16x32): lane L row M = L&15 ;
//   halves 0..7  -> K = kbase+0..7,  halves 8..15 -> K = kbase+16..23
//   where kbase = kt*32 + (L>=16 ? 8 : 0).
// ---------------------------------------------------------------------------
__global__ void gemm_wmma(const float* __restrict__ A, int lda, int ktiles,
                          const unsigned* __restrict__ wfrag,
                          float* __restrict__ out, int nrows) {
    __shared__ __attribute__((aligned(32))) unsigned sfrag[4 * 8 * 256]; // 32 KB max

    // Stage all B fragments to LDS (all threads participate before any exit).
    int nwords = ktiles * 8 * 256;
    for (int i = threadIdx.x; i < nwords; i += blockDim.x) sfrag[i] = wfrag[i];
    __syncthreads();

    int wave = blockIdx.x * (blockDim.x >> 5) + (threadIdx.x >> 5);
    int m0 = wave * 32;                            // two 16-row tiles per wave
    if (m0 >= nrows) return;                       // wave-uniform: EXEC stays full
    int lane = threadIdx.x & 31;
    int row  = m0 + (lane & 15);
    int hiK  = (lane >> 4) << 3;                   // +8 for lanes 16..31

    v8f acc0[8], acc1[8];
    v8f zero = {0.f, 0.f, 0.f, 0.f, 0.f, 0.f, 0.f, 0.f};
#pragma unroll
    for (int nt = 0; nt < 8; ++nt) { acc0[nt] = zero; acc1[nt] = zero; }

    const float* arow0 = A + (long)row * lda;
    const float* arow1 = arow0 + (long)16 * lda;
    for (int kt = 0; kt < ktiles; ++kt) {
        int kb = kt * 32 + hiK;
        if (kt + 1 < ktiles) {                      // prefetch next A K-tile
            __builtin_prefetch(arow0 + kb + 32, 0, 3);
            __builtin_prefetch(arow1 + kb + 32, 0, 3);
        }
        v16bf av0 = load_a_frag(arow0, kb);
        v16bf av1 = load_a_frag(arow1, kb);

        const unsigned* fb = sfrag + kt * 2048 + lane * 8;
#pragma unroll
        for (int nt = 0; nt < 8; ++nt) {
            v8u bu = *(const v8u*)(fb + nt * 256);  // ds_load_b128 x2
            v16bf bv = __builtin_bit_cast(v16bf, bu);
            acc0[nt] = __builtin_amdgcn_wmma_f32_16x16x32_bf16(
                false, av0, false, bv, (short)0, acc0[nt], false, false);
            acc1[nt] = __builtin_amdgcn_wmma_f32_16x16x32_bf16(
                false, av1, false, bv, (short)0, acc1[nt], false, false);
        }
    }

    // D layout: VGPR r, lane L -> (M = r + (L>=16 ? 8:0), N = L&15)
    int col0  = lane & 15;
    int rbase = m0 + ((lane >> 4) << 3);
#pragma unroll
    for (int nt = 0; nt < 8; ++nt)
#pragma unroll
        for (int r = 0; r < 8; ++r) {
            out[(long)(rbase + r) * HID + nt * 16 + col0]      = acc0[nt][r];
            out[(long)(rbase + 16 + r) * HID + nt * 16 + col0] = acc1[nt][r];
        }
}

// es/ed per (node, head)
__global__ void scores_k(const float* __restrict__ h,
                         const float* __restrict__ asrc,
                         const float* __restrict__ adst,
                         float* __restrict__ es, float* __restrict__ ed) {
    int idx = blockIdx.x * blockDim.x + threadIdx.x;
    if (idx >= NN * HEADS) return;
    int hd = idx & 3, n = idx >> 2;
    const float* hp = h + (long)n * HID + hd * FILTERS;
    float s = 0.f, d = 0.f;
#pragma unroll 8
    for (int f = 0; f < FILTERS; ++f) {
        float v = hp[f];
        s += v * asrc[hd * FILTERS + f];
        d += v * adst[hd * FILTERS + f];
    }
    es[idx] = s;
    ed[idx] = d;
}

__global__ void init_k(unsigned* __restrict__ m, float* __restrict__ denom,
                       float* __restrict__ agg) {
    int idx = blockIdx.x * blockDim.x + threadIdx.x;
    if (idx < NN * HEADS) { m[idx] = 0x00800000u /* flip(-FLT_MAX) */; denom[idx] = 0.f; }
    if (idx < NN * HID) agg[idx] = 0.f;
}

// pass 1: leaky-relu score per (edge, head) + segment max (flipped-uint atomicMax)
__global__ void edge_pass1(const int* __restrict__ ei,
                           const float* __restrict__ es,
                           const float* __restrict__ ed,
                           float* __restrict__ ebuf, unsigned* __restrict__ m) {
    int idx = blockIdx.x * blockDim.x + threadIdx.x;
    if (idx >= EP * HEADS) return;
    int h = idx & 3, e = idx >> 2;
    int s = (e < EE) ? ei[e]      : (e - EE);
    int d = (e < EE) ? ei[EE + e] : (e - EE);
    float v = es[s * HEADS + h] + ed[d * HEADS + h];
    v = (v > 0.f) ? v : LEAKY * v;
    ebuf[idx] = v;
    atomicMax(&m[d * HEADS + h], flipf(v));
}

// pass 2: ex = exp(e - max); segment sum into denom
__global__ void edge_pass2(const int* __restrict__ ei,
                           float* __restrict__ ebuf,
                           const unsigned* __restrict__ m,
                           float* __restrict__ denom) {
    int idx = blockIdx.x * blockDim.x + threadIdx.x;
    if (idx >= EP * HEADS) return;
    int h = idx & 3, e = idx >> 2;
    int d = (e < EE) ? ei[EE + e] : (e - EE);
    float ex = __expf(ebuf[idx] - unflipf(m[d * HEADS + h]));
    ebuf[idx] = ex;
    atomicAdd(&denom[d * HEADS + h], ex);
}

// pass 3: agg[dst] += h[src] * alpha  (128 features / edge, 2 edges / block)
__global__ void edge_pass3(const int* __restrict__ ei,
                           const float* __restrict__ ebuf,
                           const float* __restrict__ denom,
                           const float* __restrict__ h,
                           float* __restrict__ agg) {
    int e = blockIdx.x * 2 + (threadIdx.x >> 7);
    if (e >= EP) return;
    int f = threadIdx.x & 127;
    int s = (e < EE) ? ei[e]      : (e - EE);
    int d = (e < EE) ? ei[EE + e] : (e - EE);
    int hd = f >> 5;
    float alpha = ebuf[e * HEADS + hd] / (denom[d * HEADS + hd] + 1e-16f);
    atomicAdd(&agg[(long)d * HID + f], h[(long)s * HID + f] * alpha);
}

// out = elu(agg + bias)
__global__ void finalize_k(const float* __restrict__ agg,
                           const float* __restrict__ bias,
                           float* __restrict__ out) {
    int idx = blockIdx.x * blockDim.x + threadIdx.x;
    if (idx >= NN * HID) return;
    float v = agg[idx] + bias[idx & (HID - 1)];
    out[idx] = (v > 0.f) ? v : (__expf(v) - 1.f);
}

// y = relu(h @ lw1^T + lb1) @ lw2^T + lb2
__global__ void mlp_k(const float* __restrict__ h,
                      const float* __restrict__ lw1, const float* __restrict__ lb1,
                      const float* __restrict__ lw2, const float* __restrict__ lb2,
                      float* __restrict__ y) {
    int n = blockIdx.x * blockDim.x + threadIdx.x;
    if (n >= NN) return;
    const float* hr = h + (long)n * HID;
    float acc = lb2[0];
    for (int o = 0; o < FILTERS; ++o) {
        float s = lb1[o];
        const float* w = lw1 + o * HID;
#pragma unroll 8
        for (int k = 0; k < HID; ++k) s += hr[k] * w[k];
        acc += fmaxf(s, 0.f) * lw2[o];
    }
    y[n] = acc;
}

extern "C" void kernel_launch(void* const* d_in, const int* in_sizes, int n_in,
                              void* d_out, int out_size, void* d_ws, size_t ws_size,
                              hipStream_t stream) {
    const float* x   = (const float*)d_in[0];   // [N, T, FIN]
    const int*   ei  = (const int*)  d_in[1];   // [2, E]
    const float* W1  = (const float*)d_in[2];
    const float* as1 = (const float*)d_in[3];
    const float* ad1 = (const float*)d_in[4];
    const float* b1  = (const float*)d_in[5];
    const float* W2  = (const float*)d_in[6];
    const float* as2 = (const float*)d_in[7];
    const float* ad2 = (const float*)d_in[8];
    const float* b2  = (const float*)d_in[9];
    const float* lw1 = (const float*)d_in[10];
    const float* lb1 = (const float*)d_in[11];
    const float* lw2 = (const float*)d_in[12];
    const float* lb2 = (const float*)d_in[13];
    float* out = (float*)d_out;                 // [T, N, 1]

    float* ws   = (float*)d_ws;
    float* bufA = ws;                           // N*HID
    float* bufB = bufA + (size_t)NN * HID;      // N*HID
    float* es   = bufB + (size_t)NN * HID;      // N*HEADS
    float* ed   = es + NN * HEADS;              // N*HEADS
    unsigned* m = (unsigned*)(ed + NN * HEADS); // N*HEADS
    float* denom = (float*)(m + NN * HEADS);    // N*HEADS
    float* ebuf  = denom + NN * HEADS;          // EP*HEADS
    unsigned* wf1 = (unsigned*)(ebuf + (size_t)EP * HEADS); // 2*8*256
    unsigned* wf2 = wf1 + 2 * 8 * 256;                      // 4*8*256

    const int B = 256;
    const int gPack1 = (2 * 8 * 256 + B - 1) / B;
    const int gPack2 = (4 * 8 * 256 + B - 1) / B;
    const int gGemm  = (NN / 32 + 7) / 8;            // 8 waves (32 rows each) per block
    const int gNH    = (NN * HEADS + B - 1) / B;
    const int gInit  = (NN * HID + B - 1) / B;
    const int gEdgeH = (EP * HEADS + B - 1) / B;
    const int gEdgeF = EP / 2;                       // 2 edges x 128 features / block
    const int gNode  = (NN + B - 1) / B;

    // Weight fragments (deterministic; inputs are never mutated)
    pack_w<<<gPack1, B, 0, stream>>>(W1, FIN, wf1);
    pack_w<<<gPack2, B, 0, stream>>>(W2, HID, wf2);

    for (int t = 0; t < TT; ++t) {
        // ---- GAT layer 1: input x[:, t, :] (stride T*FIN), h -> bufA
        gemm_wmma<<<gGemm, B, 0, stream>>>(x + (size_t)t * FIN, TT * FIN, FIN / 32,
                                           wf1, bufA, NN);
        scores_k<<<gNH, B, 0, stream>>>(bufA, as1, ad1, es, ed);
        init_k<<<gInit, B, 0, stream>>>(m, denom, bufB);
        edge_pass1<<<gEdgeH, B, 0, stream>>>(ei, es, ed, ebuf, m);
        edge_pass2<<<gEdgeH, B, 0, stream>>>(ei, ebuf, m, denom);
        edge_pass3<<<gEdgeF, B, 0, stream>>>(ei, ebuf, denom, bufA, bufB);
        finalize_k<<<gInit, B, 0, stream>>>(bufB, b1, bufA);   // bufA = elu(layer1)

        // ---- GAT layer 2: input bufA, h -> bufB
        gemm_wmma<<<gGemm, B, 0, stream>>>(bufA, HID, HID / 32, wf2, bufB, NN);
        scores_k<<<gNH, B, 0, stream>>>(bufB, as2, ad2, es, ed);
        init_k<<<gInit, B, 0, stream>>>(m, denom, bufA);
        edge_pass1<<<gEdgeH, B, 0, stream>>>(ei, es, ed, ebuf, m);
        edge_pass2<<<gEdgeH, B, 0, stream>>>(ei, ebuf, m, denom);
        edge_pass3<<<gEdgeF, B, 0, stream>>>(ei, ebuf, denom, bufB, bufA);
        finalize_k<<<gInit, B, 0, stream>>>(bufA, b2, bufB);   // bufB = elu(layer2)

        // ---- MLP head -> out[t, :, 0]
        mlp_k<<<gNode, B, 0, stream>>>(bufB, lw1, lb1, lw2, lb2, out + (size_t)t * NN);
    }
}
